// VerbAnticipator_6476810682694
// MI455X (gfx1250) — matrix operand order repaired and checked by reading, not compiled, not run on hardware
//
#include <hip/hip_runtime.h>
#include <hip/hip_bf16.h>

typedef __attribute__((ext_vector_type(16))) _Float16 v16h;
typedef __attribute__((ext_vector_type(8)))  float    v8f;
typedef __attribute__((ext_vector_type(4)))  float    f4;

#define DEV __device__ __forceinline__

DEV float sigm(float x) { return 1.0f / (1.0f + expf(-x)); }

// ---------------- max-pool over 4096 rows of [4096,1024] (single-pass: NT) ----
__global__ void k_maxpool_partial(const float* __restrict__ in, float* __restrict__ part) {
    int col = blockIdx.x * 256 + threadIdx.x;      // 0..1023
    size_t rbase = (size_t)blockIdx.y * 256;
    const float* p = in + rbase * 1024 + col;
    float m = -3.4e38f;
    for (int r = 0; r < 256; ++r)
        m = fmaxf(m, __builtin_nontemporal_load(p + (size_t)r * 1024));
    part[blockIdx.y * 1024 + col] = m;
}
__global__ void k_maxpool_final(const float* __restrict__ part, float* __restrict__ mf) {
    int col = blockIdx.x * 256 + threadIdx.x;
    float m = part[col];
    for (int j = 1; j < 16; ++j) m = fmaxf(m, part[j * 1024 + col]);
    mf[col] = m;
}

// ---------------- wave-per-row GEMV: y = W@x (+b0)(+b1)(+add) ----------------
// nt=1: stream W with non-temporal loads (single-use weights; keep L2 for the
// 20x-reused rW_hh / 2x-reused rW_ih & Wp; unique WS ~196MB vs 192MB L2).
__global__ void k_gemv(const float* __restrict__ W, const float* __restrict__ x,
                       const float* __restrict__ b0, const float* __restrict__ b1,
                       const float* __restrict__ add,
                       float* __restrict__ y, int M, int K, int nt) {
    int row  = (int)((blockIdx.x * blockDim.x + threadIdx.x) >> 5);
    int lane = threadIdx.x & 31;
    if (row >= M) return;
    const float* wr = W + (size_t)row * K;
    float s = 0.f;
    if (nt) {
        for (int k = lane * 4; k < K; k += 128) {
            f4 w = __builtin_nontemporal_load((const f4*)(wr + k));
            f4 v = *(const f4*)(x + k);
            s = fmaf(w.x, v.x, s); s = fmaf(w.y, v.y, s);
            s = fmaf(w.z, v.z, s); s = fmaf(w.w, v.w, s);
        }
    } else {
        for (int k = lane * 4; k < K; k += 128) {
            f4 w = *(const f4*)(wr + k);
            f4 v = *(const f4*)(x + k);
            s = fmaf(w.x, v.x, s); s = fmaf(w.y, v.y, s);
            s = fmaf(w.z, v.z, s); s = fmaf(w.w, v.w, s);
        }
    }
    for (int off = 16; off > 0; off >>= 1) s += __shfl_down(s, off, 32);
    if (lane == 0) {
        if (b0)  s += b0[row];
        if (b1)  s += b1[row];
        if (add) s += add[row];
        y[row] = s;
    }
}

// ---------------- small helpers ----------------
__global__ void k_pack2(const float* a, const float* b, float* __restrict__ dst) {
    int i = blockIdx.x * 256 + threadIdx.x;        // 0..2047
    if (i < 1024) dst[i] = a ? a[i] : 0.f;
    else          dst[i] = b ? b[i - 1024] : 0.f;
}
__global__ void k_zero(float* __restrict__ p, int n) {
    int i = blockIdx.x * 256 + threadIdx.x;
    if (i < n) p[i] = 0.f;
}
// LSTM cell with zero initial h,c (goal LSTM): h = sig(o)*tanh(sig(i)*tanh(g))
__global__ void k_lstm0(const float* __restrict__ g4, float* __restrict__ h) {
    int n = blockIdx.x * 256 + threadIdx.x;        // 0..1023
    float i  = sigm(g4[n]);
    float gg = tanhf(g4[2048 + n]);
    float o  = sigm(g4[3072 + n]);
    h[n] = o * tanhf(i * gg);
}
// recurrent LSTM step: gates already contain x@Wih.T + h@Whh.T + rb
__global__ void k_rstep(const float* __restrict__ g4, float* __restrict__ cc,
                        float* __restrict__ hh, float* __restrict__ aout) {
    int n = blockIdx.x * 256 + threadIdx.x;        // 0..1023
    float i  = sigm(g4[n]);
    float f  = sigm(g4[1024 + n]);
    float gg = tanhf(g4[2048 + n]);
    float o  = sigm(g4[3072 + n]);
    float c  = f * cc[n] + i * gg;
    cc[n] = c;
    float h = o * tanhf(c);
    hh[n] = h;
    aout[n] = h;
}

// bd = sum((fs-g)^2)/1024 ; bf = fs
__global__ void k_init_best(const float* __restrict__ fs, const float* __restrict__ g,
                            float* __restrict__ bd, float* __restrict__ bf) {
    __shared__ float red[256];
    int t = threadIdx.x;
    float s = 0.f;
    for (int i = t; i < 1024; i += 256) { float d = fs[i] - g[i]; s += d * d; bf[i] = fs[i]; }
    red[t] = s; __syncthreads();
    for (int o = 128; o > 0; o >>= 1) { if (t < o) red[t] += red[t + o]; __syncthreads(); }
    if (t == 0) bd[0] = red[0] * (1.0f / 1024.0f);
}

// sequential greedy score scan over 10 candidates
__global__ void k_score(const float* __restrict__ nfs, const float* __restrict__ g,
                        float* __restrict__ bd, float* __restrict__ bf,
                        float* __restrict__ sc) {
    __shared__ float red[256];
    __shared__ float dj[10];
    __shared__ int   bestj;
    int t = threadIdx.x;
    for (int j = 0; j < 10; ++j) {
        const float* r = nfs + j * 1024;
        float s = 0.f;
        for (int i = t; i < 1024; i += 256) { float d = r[i] - g[i]; s += d * d; }
        red[t] = s; __syncthreads();
        for (int o = 128; o > 0; o >>= 1) { if (t < o) red[t] += red[t + o]; __syncthreads(); }
        if (t == 0) dj[j] = red[0] * (1.0f / 1024.0f);
        __syncthreads();
    }
    if (t == 0) {
        float b = bd[0]; int bj = -1;
        for (int j = 0; j < 10; ++j) {
            sc[j] = b - dj[j];
            if (dj[j] < b) { b = dj[j]; bj = j; }
        }
        bd[0] = b; bestj = bj;
    }
    __syncthreads();
    if (bestj >= 0) {
        const float* r = nfs + bestj * 1024;
        for (int i = t; i < 1024; i += 256) bf[i] = r[i];
    }
}

// i* = argmin(s1) (first min); a2in=a1[i*], f2in=f1[i*] (also to d_out)
__global__ void k_pick1(const float* __restrict__ s1, const float* __restrict__ a1,
                        const float* __restrict__ f1, float* __restrict__ a2in,
                        float* __restrict__ f2in, int* __restrict__ istar,
                        float* __restrict__ out_f2in) {
    __shared__ int js;
    int t = threadIdx.x;
    if (t == 0) {
        int bi = 0; float bv = s1[0];
        for (int j = 1; j < 10; ++j) if (s1[j] < bv) { bv = s1[j]; bi = j; }
        js = bi; istar[0] = bi;
    }
    __syncthreads();
    const float* ar = a1 + js * 1024;
    const float* fr = f1 + js * 1024;
    for (int i = t; i < 1024; i += 256) {
        a2in[i] = ar[i]; f2in[i] = fr[i]; out_f2in[i] = fr[i];
    }
}

// X[16,3072] = rows 0..9: concat(f_in, a_js[j], g); rows 10..15: 0
__global__ void k_buildX(const float* __restrict__ fin, const float* __restrict__ ajs,
                         const float* __restrict__ g, float* __restrict__ X) {
    int idx = blockIdx.x * 256 + threadIdx.x;      // < 49152
    int m = idx / 3072, k = idx - m * 3072;
    float v = 0.f;
    if (m < 10) {
        if (k < 1024)       v = fin[k];
        else if (k < 2048)  v = ajs[m * 1024 + (k - 1024)];
        else                v = g[k - 2048];
    }
    X[idx] = v;
}

// pick 2 lowest scores (popped node removed), build chain means, head input P
__global__ void k_final(const float* __restrict__ s1, const float* __restrict__ s2,
                        const int* __restrict__ istar,
                        const float* __restrict__ a1, const float* __restrict__ f1,
                        const float* __restrict__ a2, const float* __restrict__ f2,
                        const float* __restrict__ a2in, const float* __restrict__ f2in,
                        const float* __restrict__ fs, const float* __restrict__ g,
                        float* __restrict__ P, float* __restrict__ out_pfeat,
                        float* __restrict__ out_pgoal) {
    __shared__ int idx[2];
    int t = threadIdx.x;
    if (t == 0) {
        float s[20];
        int is = istar[0];
        for (int j = 0; j < 10; ++j) s[j] = (j == is) ? 3.0e38f : s1[j];
        for (int j = 0; j < 10; ++j) s[10 + j] = s2[j];
        int i0 = 0;
        for (int j = 1; j < 20; ++j) if (s[j] < s[i0]) i0 = j;
        int i1 = -1;
        for (int j = 0; j < 20; ++j) {
            if (j == i0) continue;
            if (i1 < 0 || s[j] < s[i1]) i1 = j;
        }
        idx[0] = i0; idx[1] = i1;
    }
    __syncthreads();
    for (int c = t; c < 1024; c += 256) {
        P[c] = fs[c];
        for (int m = 3; m < 16; ++m) P[m * 1024 + c] = 0.f;
        for (int r = 0; r < 2; ++r) {
            int id = idx[r];
            float pav, pfv;
            if (id < 10) {
                pav = a1[id * 1024 + c] * 0.5f;
                pfv = (fs[c] + f1[id * 1024 + c]) * 0.5f;
            } else {
                int j = id - 10;
                pav = (a2in[c] + a2[j * 1024 + c]) * (1.0f / 3.0f);
                pfv = (fs[c] + f2in[c] + f2[j * 1024 + c]) * (1.0f / 3.0f);
            }
            P[(1 + r) * 1024 + c] = pav;
            out_pfeat[r * 1024 + c] = pfv;
            out_pgoal[r * 1024 + c] = g[c];
        }
    }
}

// ---------------- WMMA fragment loads (f32 -> f16), ISA 7.12.2 layouts --------
// A 16x32 (MxK) f16: lanes 0-15 m, K 0..7 & 16..23; lanes 16-31 m, K 8..15 & 24..31
DEV v16h loadA(const float* __restrict__ A, int lda, int k0, int lane) {
    int m = lane & 15;
    const float* p = A + (size_t)m * lda + k0 + ((lane < 16) ? 0 : 8);
    v16h a;
#pragma unroll
    for (int h = 0; h < 8; ++h) a[h] = (_Float16)p[h];
#pragma unroll
    for (int h = 0; h < 8; ++h) a[8 + h] = (_Float16)p[16 + h];
    return a;
}
// B 32x16 (KxN), B[k][n] = W[n][k] (W row-major [N,K]): lanes 0-15 K 0..7 & 16..23; lanes 16-31 K 8..15 & 24..31
DEV v16h loadB(const float* __restrict__ W, int ldw, int n0, int N, int k0, int lane) {
    int n = n0 + (lane & 15);
    v16h b;
    if (n >= N) {
#pragma unroll
        for (int h = 0; h < 16; ++h) b[h] = (_Float16)0.0f;
        return b;
    }
    const float* p = W + (size_t)n * ldw + k0 + ((lane < 16) ? 0 : 8);
#pragma unroll
    for (int h = 0; h < 8; ++h) b[h] = (_Float16)p[h];
#pragma unroll
    for (int h = 0; h < 8; ++h) b[8 + h] = (_Float16)p[16 + h];
    return b;
}

// nfs[10,1024] = relu(X[16,3072] @ Wp^T + bp), Wp row-major [1024,3072]
__global__ void k_wmma_featpred(const float* __restrict__ X, const float* __restrict__ Wp,
                                const float* __restrict__ bp, float* __restrict__ nfs) {
    int wave = threadIdx.x >> 5, lane = threadIdx.x & 31;
    int n0 = blockIdx.x * 64 + wave * 16;
    v8f c;
#pragma unroll
    for (int r = 0; r < 8; ++r) c[r] = 0.f;
    for (int k0 = 0; k0 < 3072; k0 += 32) {
        v16h a = loadA(X, 3072, k0, lane);
        v16h b = loadB(Wp, 3072, n0, 1024, k0, lane);
        c = __builtin_amdgcn_wmma_f32_16x16x32_f16(false, a, false, b, (short)0, c, false, false);
    }
    int n = n0 + (lane & 15);
#pragma unroll
    for (int r = 0; r < 8; ++r) {
        int m = (lane < 16) ? r : (r + 8);
        if (m < 10) nfs[m * 1024 + n] = fmaxf(c[r] + bp[n], 0.f);
    }
}

// heads: P[16,1024] @ W^T + bias; row0 -> cur (optional), rows1-2 -> out2[2,N]
__global__ void k_wmma_heads(const float* __restrict__ P, const float* __restrict__ W,
                             const float* __restrict__ bias, float* __restrict__ out2,
                             float* __restrict__ cur, int N) {
    int wave = threadIdx.x >> 5, lane = threadIdx.x & 31;
    int n0 = blockIdx.x * 64 + wave * 16;
    v8f c;
#pragma unroll
    for (int r = 0; r < 8; ++r) c[r] = 0.f;
    for (int k0 = 0; k0 < 1024; k0 += 32) {
        v16h a = loadA(P, 1024, k0, lane);
        v16h b = loadB(W, 1024, n0, N, k0, lane);
        c = __builtin_amdgcn_wmma_f32_16x16x32_f16(false, a, false, b, (short)0, c, false, false);
    }
    int n = n0 + (lane & 15);
    if (n >= N) return;
#pragma unroll
    for (int r = 0; r < 8; ++r) {
        int m = (lane < 16) ? r : (r + 8);
        float v = c[r] + bias[n];
        if (m == 0)      { if (cur) cur[n] = v; }
        else if (m == 1) out2[n] = v;
        else if (m == 2) out2[N + n] = v;
    }
}

// ======================= host orchestration =======================
struct Ctx {
    const float *rW_ih, *rW_hh, *rb_ih, *rb_hh, *Wp, *bp;
    float *xcat, *gx, *gates, *hh, *cc, *X, *g, *bd, *bf;
};

static void run_expand(const Ctx& c, const float* a_in /*null => zeros*/, const float* f_in,
                       float* a_js, float* f_js, float* scores, hipStream_t stream) {
    k_pack2<<<8, 256, 0, stream>>>(a_in, f_in, c.xcat);                      // xin = [a_in, f_in]
    k_gemv<<<512, 256, 0, stream>>>(c.rW_ih, c.xcat, c.rb_ih, c.rb_hh, nullptr,
                                    c.gx, 4096, 2048, 0);                     // reused -> cache
    k_zero<<<8, 256, 0, stream>>>(c.hh, 2048);                                // hh,cc contiguous
    for (int j = 0; j < 10; ++j) {
        k_gemv<<<512, 256, 0, stream>>>(c.rW_hh, c.hh, nullptr, nullptr, c.gx,
                                        c.gates, 4096, 1024, 0);              // 20x reuse -> cache
        k_rstep<<<4, 256, 0, stream>>>(c.gates, c.cc, c.hh, a_js + j * 1024);
    }
    k_buildX<<<192, 256, 0, stream>>>(f_in, a_js, c.g, c.X);
    k_wmma_featpred<<<16, 128, 0, stream>>>(c.X, c.Wp, c.bp, f_js);
    k_score<<<1, 256, 0, stream>>>(f_js, c.g, c.bd, c.bf, scores);
}

extern "C" void kernel_launch(void* const* d_in, const int* in_sizes, int n_in,
                              void* d_out, int out_size, void* d_ws, size_t ws_size,
                              hipStream_t stream) {
    (void)in_sizes; (void)n_in; (void)out_size; (void)ws_size;
    const float* tsn    = (const float*)d_in[0];
    const float* W_fe   = (const float*)d_in[1];
    const float* b_fe   = (const float*)d_in[2];
    const float* gW_ih0 = (const float*)d_in[3];
    const float* gW_ihR = (const float*)d_in[4];
    // d_in[5] gW_hh is multiplied by h==0 in the reference -> unused
    const float* gb_ih  = (const float*)d_in[6];
    const float* gb_hh  = (const float*)d_in[7];
    const float* rW_ih  = (const float*)d_in[8];
    const float* rW_hh  = (const float*)d_in[9];
    const float* rb_ih  = (const float*)d_in[10];
    const float* rb_hh  = (const float*)d_in[11];
    const float* Wp     = (const float*)d_in[12];
    const float* bp     = (const float*)d_in[13];
    const float* We2a   = (const float*)d_in[14];
    const float* be2a   = (const float*)d_in[15];
    const float* We2v   = (const float*)d_in[16];
    const float* be2v   = (const float*)d_in[17];
    const float* We2n   = (const float*)d_in[18];
    const float* be2n   = (const float*)d_in[19];

    float* out = (float*)d_out;
    float* o_pa    = out;            // [2,2513]
    float* o_pv    = out + 5026;     // [2,125]
    float* o_pn    = out + 5276;     // [2,352]
    float* o_pfeat = out + 5980;     // [2,1024]
    float* o_pgoal = out + 8028;     // [2,1024]
    float* o_f2in  = out + 10076;    // [1,1024]
    float* o_cur   = out + 11100;    // [1,2513]

    float* wsp = (float*)d_ws;
    auto alloc = [&](size_t n) { float* p = wsp; wsp += n; return p; };
    float* mf      = alloc(1024);
    float* fs      = alloc(1024);
    float* g       = alloc(1024);
    float* h       = alloc(1024);
    float* xcat    = alloc(2048);
    float* gates   = alloc(4096);
    float* gx      = alloc(4096);
    float* hh      = alloc(1024);
    float* cc      = alloc(1024);   // must stay contiguous right after hh
    float* a1      = alloc(10240);
    float* f1      = alloc(10240);
    float* a2      = alloc(10240);
    float* f2      = alloc(10240);
    float* a2in    = alloc(1024);
    float* f2in    = alloc(1024);
    float* bf      = alloc(1024);
    float* bd      = alloc(4);
    float* s1      = alloc(16);
    float* s2      = alloc(16);
    int*   istar   = (int*)alloc(16);
    float* X       = alloc(16 * 3072);
    float* P       = alloc(16 * 1024);
    float* partial = alloc(16 * 1024);

    // 1. fs = maxpool(tsn) @ W_fe.T + b_fe   (tsn, W_fe single-pass -> NT)
    k_maxpool_partial<<<dim3(4, 16), 256, 0, stream>>>(tsn, partial);
    k_maxpool_final<<<4, 256, 0, stream>>>(partial, mf);
    k_gemv<<<128, 256, 0, stream>>>(W_fe, mf, b_fe, nullptr, nullptr, fs, 1024, 1024, 1);

    // 2. goal LSTM (all hidden/cell states are zero => gW_hh terms vanish)
    //    gW_ih0/gW_ihR read exactly once -> NT streaming loads
    k_pack2<<<8, 256, 0, stream>>>(fs, nullptr, xcat);                       // x0 = [fs, 0]
    k_gemv<<<512, 256, 0, stream>>>(gW_ih0, xcat, gb_ih, gb_hh, nullptr, gates, 4096, 2048, 1);
    k_lstm0<<<4, 256, 0, stream>>>(gates, h);
    for (int k = 0; k < 4; ++k) {
        k_gemv<<<512, 256, 0, stream>>>(gW_ihR + (size_t)k * 4096 * 1024, h,
                                        gb_ih + (k + 1) * 4096, gb_hh + (k + 1) * 4096,
                                        nullptr, gates, 4096, 1024, 1);
        k_lstm0<<<4, 256, 0, stream>>>(gates, (k == 3) ? g : h);
    }

    // 3. expansions
    Ctx c{rW_ih, rW_hh, rb_ih, rb_hh, Wp, bp, xcat, gx, gates, hh, cc, X, g, bd, bf};
    k_init_best<<<1, 256, 0, stream>>>(fs, g, bd, bf);
    run_expand(c, nullptr, fs, a1, f1, s1, stream);                          // expansion 1
    k_pick1<<<1, 256, 0, stream>>>(s1, a1, f1, a2in, f2in, istar, o_f2in);
    run_expand(c, a2in, f2in, a2, f2, s2, stream);                           // expansion 2

    // 4. select 2 best, build chain means + head input P (row0=fs, rows1-2=pa)
    k_final<<<1, 256, 0, stream>>>(s1, s2, istar, a1, f1, a2, f2, a2in, f2in,
                                   fs, g, P, o_pfeat, o_pgoal);

    // 5. classifier heads via WMMA (cur_action folded into row 0 of We2a pass)
    k_wmma_heads<<<40, 128, 0, stream>>>(P, We2a, be2a, o_pa, o_cur, 2513);
    k_wmma_heads<<<2, 128, 0, stream>>>(P, We2v, be2v, o_pv, nullptr, 125);
    k_wmma_heads<<<6, 128, 0, stream>>>(P, We2n, be2n, o_pn, nullptr, 352);
}
